// DeepFM_33595234190062
// MI455X (gfx1250) — matrix operand order repaired and checked by reading, not compiled, not run on hardware
//
#include <hip/hip_runtime.h>

// ---------------------------------------------------------------------------
// DeepFM fused forward for MI455X (gfx1250, wave32, WMMA).
//  Phase 1: embedding gather (f32) + FM sums + linear term, store bf16 A in LDS
//  Phase 2: GEMM1 [64x1248]@[1248x256] via v_wmma_f32_16x16x32_bf16
//  Phase 3: GEMM2 [64x256]@[256x128], Phase 4: GEMM3 [64x128]@[128x64]
//  Phase 5: 66-wide head -> logit1, logit2
// ---------------------------------------------------------------------------

#define B_SZ   16384
#define F_CAT  26
#define VOCAB  100000
#define N_NUM  13
#define D_EMB  32
#define K1     1248         // (26+13)*32
#define H1     256
#define H2     128
#define H3     64

#define TILE_M 64
#define LDA    1256         // hA stride in halves (1248 + 8 pad -> conflict-free b128)
#define LDW    40           // staged weight slab stride in halves (32 + 8 pad)
#define LDH1   264          // h1 stride in halves
#define LDH2   136          // h2 stride in halves
#define LDH3   68           // h3 stride in floats

typedef __attribute__((ext_vector_type(16))) __bf16        v16bf;
typedef __attribute__((ext_vector_type(8)))  float          v8f;
typedef __attribute__((ext_vector_type(16))) unsigned short v16u;
typedef __attribute__((ext_vector_type(8)))  unsigned short v8u;

// LDS layout (bytes)
#define OFF_HA   0
#define SZ_HA    (TILE_M * LDA * 2)       // 160768
#define OFF_WB   (OFF_HA + SZ_HA)
#define SZ_WB    (256 * LDW * 2)          // 20480
#define OFF_H1   (OFF_WB + SZ_WB)
#define SZ_H1    (TILE_M * LDH1 * 2)      // 33792
#define OFF_H2   (OFF_H1 + SZ_H1)
#define SZ_H2    (TILE_M * LDH2 * 2)      // 17408
#define OFF_H3   (OFF_H2 + SZ_H2)
#define SZ_H3    (TILE_M * LDH3 * 4)      // 17408
#define OFF_FM   (OFF_H3 + SZ_H3)
#define SZ_FM    (TILE_M * 4 * 4)
#define OFF_LIN  (OFF_FM + SZ_FM)
#define SZ_LIN   (TILE_M * 4 * 4)
#define LDS_BYTES (OFF_LIN + SZ_LIN)      // 251904 (< 320KB/WGP)

__device__ __forceinline__ unsigned short f2bf(float x) {
  unsigned int u = __float_as_uint(x);           // round-to-nearest-even
  return (unsigned short)((u + 0x7FFFu + ((u >> 16) & 1u)) >> 16);
}

// Build a v16bf A/B fragment from two contiguous 8-half (16B) LDS chunks.
__device__ __forceinline__ v16bf frag16(const unsigned short* p0,
                                        const unsigned short* p1) {
  v8u lo = *(const v8u*)p0;
  v8u hi = *(const v8u*)p1;
  v16u t;
#pragma unroll
  for (int i = 0; i < 8; ++i) { t[i] = lo[i]; t[i + 8] = hi[i]; }
  return __builtin_bit_cast(v16bf, t);
}

extern "C" __global__ __launch_bounds__(256)
void deepfm_fused(const int*   __restrict__ x_cat,
                  const float* __restrict__ x_num,
                  const float* __restrict__ emb0,
                  const float* __restrict__ emb1,
                  const float* __restrict__ num_bias,
                  const float* __restrict__ num_emb,
                  const float* __restrict__ W1, const float* __restrict__ b1,
                  const float* __restrict__ W2, const float* __restrict__ b2,
                  const float* __restrict__ W3, const float* __restrict__ b3,
                  const float* __restrict__ Wc, const float* __restrict__ bc,
                  const float* __restrict__ Wo, const float* __restrict__ bo,
                  float* __restrict__ out) {
  extern __shared__ unsigned char smem_raw[];
  unsigned short* hA  = (unsigned short*)(smem_raw + OFF_HA);
  unsigned short* wB  = (unsigned short*)(smem_raw + OFF_WB);
  unsigned short* h1  = (unsigned short*)(smem_raw + OFF_H1);
  unsigned short* h2  = (unsigned short*)(smem_raw + OFF_H2);
  float*          h3  = (float*)(smem_raw + OFF_H3);
  float*          fmp = (float*)(smem_raw + OFF_FM);
  float*          lnp = (float*)(smem_raw + OFF_LIN);

  const int tid  = threadIdx.x;
  const int lane = tid & 31;
  const int wave = tid >> 5;
  const int b0   = blockIdx.x * TILE_M;

  // ---------------- Phase 1: gather + FM + linear --------------------------
  {
    const int r   = tid >> 2;                 // row in tile: 0..63
    const int sub = tid & 3;                  // owns dims d = sub*8 .. sub*8+7
    const int b   = b0 + r;
    float s[8], q[8];
#pragma unroll
    for (int d = 0; d < 8; ++d) { s[d] = 0.f; q[d] = 0.f; }
    float lp = 0.f;
    unsigned short* hrow = hA + r * LDA;

#pragma unroll 1
    for (int f = 0; f < F_CAT; ++f) {
      int v = x_cat[b * F_CAT + f];
      const float* e = emb1 + ((size_t)(f * VOCAB + v)) * D_EMB + sub * 8;
      v8u o;
#pragma unroll
      for (int d = 0; d < 8; ++d) {
        float ev = e[d];
        s[d] += ev; q[d] += ev * ev;
        o[d] = f2bf(ev);
      }
      *(v8u*)(hrow + f * D_EMB + sub * 8) = o;
      if ((f & 3) == sub) lp += emb0[f * VOCAB + v];
    }
#pragma unroll 1
    for (int j = 0; j < N_NUM; ++j) {
      float xv = x_num[b * N_NUM + j];
      const float* e = num_emb + j * D_EMB + sub * 8;
      v8u o;
#pragma unroll
      for (int d = 0; d < 8; ++d) {
        float ev = xv * e[d];
        s[d] += ev; q[d] += ev * ev;
        o[d] = f2bf(ev);
      }
      *(v8u*)(hrow + (F_CAT + j) * D_EMB + sub * 8) = o;
      if ((j & 3) == sub) lp += xv * num_bias[j];
    }
    float p = 0.f;
#pragma unroll
    for (int d = 0; d < 8; ++d) p += s[d] * s[d] - q[d];
    fmp[r * 4 + sub] = p;
    lnp[r * 4 + sub] = lp;
  }
  __syncthreads();

  const int aoff = (lane >= 16) ? 8  : 0;     // A-frag K offset (ISA 7.12.2)
  const int boff = (lane >= 16) ? 8  : 0;     // B-frag K offset
  const int mlo  = lane & 15;
  const int crow = (lane >= 16) ? 8 : 0;      // C/D M offset
  const v8f vzero = {};

  // ---------------- GEMM1: hA[64x1248] @ W1[1248x256] ----------------------
  {
    const int mt2 = (wave >> 2) * 2;          // two m-tiles: mt2, mt2+1
    const int ntb = (wave & 3) * 4;           // four n-tiles: ntb..ntb+3
    v8f acc[2][4];
#pragma unroll
    for (int t = 0; t < 2; ++t)
#pragma unroll
      for (int j = 0; j < 4; ++j) acc[t][j] = vzero;

    for (int k0 = 0; k0 < K1; k0 += 32) {
      // stage W1[k0:k0+32, :] transposed -> wB[n*LDW + kk] (bf16)
#pragma unroll 1
      for (int idx = tid; idx < 32 * 256; idx += 256) {
        int kk = idx >> 8, n = idx & 255;
        wB[n * LDW + kk] = f2bf(W1[(size_t)(k0 + kk) * 256 + n]);
      }
      if (k0 + 32 < K1)  // pull next slab toward L2/WGP$ (global_prefetch_b8)
        __builtin_prefetch(&W1[(size_t)(k0 + 32 + (tid >> 3)) * 256 + (tid & 7) * 32], 0, 1);
      __syncthreads();

      const unsigned short* a0p = hA + (mt2 * 16 + mlo) * LDA + k0 + aoff;
      const unsigned short* a1p = hA + ((mt2 + 1) * 16 + mlo) * LDA + k0 + aoff;
      v16bf a0 = frag16(a0p, a0p + 16);
      v16bf a1 = frag16(a1p, a1p + 16);
#pragma unroll
      for (int j = 0; j < 4; ++j) {
        const unsigned short* bp = wB + ((ntb + j) * 16 + mlo) * LDW + boff;
        v16bf bf = frag16(bp, bp + 16);
        acc[0][j] = __builtin_amdgcn_wmma_f32_16x16x32_bf16(false, a0, false, bf, (short)0, acc[0][j], false, false);
        acc[1][j] = __builtin_amdgcn_wmma_f32_16x16x32_bf16(false, a1, false, bf, (short)0, acc[1][j], false, false);
      }
      __syncthreads();
    }
#pragma unroll
    for (int j = 0; j < 4; ++j) {
      int n = (ntb + j) * 16 + mlo;
      float bias = b1[n];
#pragma unroll
      for (int t = 0; t < 2; ++t) {
        int mbase = (mt2 + t) * 16 + crow;
#pragma unroll
        for (int r = 0; r < 8; ++r) {
          float vv = acc[t][j][r] + bias;
          h2 /*dummy*/; // (no-op; keep structure)
          vv = vv > 0.f ? vv : 0.f;
          h1[(mbase + r) * LDH1 + n] = f2bf(vv);
        }
      }
    }
  }
  __syncthreads();

  // ---------------- GEMM2: h1[64x256] @ W2[256x128] ------------------------
  {
    const int mt  = wave & 3;
    const int ntb = (wave >> 2) * 4;
    v8f acc[4];
#pragma unroll
    for (int j = 0; j < 4; ++j) acc[j] = vzero;
    for (int k0 = 0; k0 < H1; k0 += 32) {
#pragma unroll 1
      for (int idx = tid; idx < 32 * 128; idx += 256) {
        int kk = idx >> 7, n = idx & 127;
        wB[n * LDW + kk] = f2bf(W2[(k0 + kk) * 128 + n]);
      }
      __syncthreads();
      const unsigned short* ap = h1 + (mt * 16 + mlo) * LDH1 + k0 + aoff;
      v16bf a0 = frag16(ap, ap + 16);
#pragma unroll
      for (int j = 0; j < 4; ++j) {
        const unsigned short* bp = wB + ((ntb + j) * 16 + mlo) * LDW + boff;
        v16bf bf = frag16(bp, bp + 16);
        acc[j] = __builtin_amdgcn_wmma_f32_16x16x32_bf16(false, a0, false, bf, (short)0, acc[j], false, false);
      }
      __syncthreads();
    }
#pragma unroll
    for (int j = 0; j < 4; ++j) {
      int n = (ntb + j) * 16 + mlo;
      float bias = b2[n];
      int mbase = mt * 16 + crow;
#pragma unroll
      for (int r = 0; r < 8; ++r) {
        float vv = acc[j][r] + bias;
        vv = vv > 0.f ? vv : 0.f;
        h2[(mbase + r) * LDH2 + n] = f2bf(vv);
      }
    }
  }
  __syncthreads();

  // ---------------- GEMM3: h2[64x128] @ W3[128x64] -------------------------
  {
    const int mt  = wave & 3;
    const int ntb = (wave >> 2) * 2;
    v8f acc[2];
#pragma unroll
    for (int j = 0; j < 2; ++j) acc[j] = vzero;
    for (int k0 = 0; k0 < H2; k0 += 32) {
#pragma unroll 1
      for (int idx = tid; idx < 32 * 64; idx += 256) {
        int kk = idx >> 6, n = idx & 63;
        wB[n * LDW + kk] = f2bf(W3[(k0 + kk) * 64 + n]);
      }
      __syncthreads();
      const unsigned short* ap = h2 + (mt * 16 + mlo) * LDH2 + k0 + aoff;
      v16bf a0 = frag16(ap, ap + 16);
#pragma unroll
      for (int j = 0; j < 2; ++j) {
        const unsigned short* bp = wB + ((ntb + j) * 16 + mlo) * LDW + boff;
        v16bf bf = frag16(bp, bp + 16);
        acc[j] = __builtin_amdgcn_wmma_f32_16x16x32_bf16(false, a0, false, bf, (short)0, acc[j], false, false);
      }
      __syncthreads();
    }
#pragma unroll
    for (int j = 0; j < 2; ++j) {
      int n = (ntb + j) * 16 + mlo;
      float bias = b3[n];
      int mbase = mt * 16 + crow;
#pragma unroll
      for (int r = 0; r < 8; ++r) {
        float vv = acc[j][r] + bias;
        vv = vv > 0.f ? vv : 0.f;
        h3[(mbase + r) * LDH3 + n] = vv;    // keep f32 for the head
      }
    }
  }
  __syncthreads();

  // ---------------- Head: concat[h3, linear, fm] @ {Wc,Wo} -----------------
  if (tid < TILE_M) {
    int r = tid;
    float lin = lnp[r * 4] + lnp[r * 4 + 1] + lnp[r * 4 + 2] + lnp[r * 4 + 3];
    float fm  = 0.5f * (fmp[r * 4] + fmp[r * 4 + 1] + fmp[r * 4 + 2] + fmp[r * 4 + 3]);
    float c1 = bc[0], c2 = bo[0];
    const float* h3r = h3 + r * LDH3;
#pragma unroll 1
    for (int n = 0; n < H3; ++n) {
      float hv = h3r[n];
      c1 += hv * Wc[n];
      c2 += hv * Wo[n];
    }
    c1 += lin * Wc[64] + fm * Wc[65];
    c2 += lin * Wo[64] + fm * Wo[65];
    int b = b0 + r;
    out[b]        = c1;   // logit1, flat [B,1]
    out[B_SZ + b] = c2;   // logit2
  }
}

extern "C" void kernel_launch(void* const* d_in, const int* in_sizes, int n_in,
                              void* d_out, int out_size, void* d_ws, size_t ws_size,
                              hipStream_t stream) {
  (void)in_sizes; (void)n_in; (void)out_size; (void)d_ws; (void)ws_size;
  const int*   x_cat    = (const int*)  d_in[0];
  const float* x_num    = (const float*)d_in[1];
  const float* emb0     = (const float*)d_in[2];
  const float* emb1     = (const float*)d_in[3];
  const float* num_bias = (const float*)d_in[4];
  const float* num_emb  = (const float*)d_in[5];
  const float* W1 = (const float*)d_in[6];
  const float* b1 = (const float*)d_in[7];
  const float* W2 = (const float*)d_in[8];
  const float* b2 = (const float*)d_in[9];
  const float* W3 = (const float*)d_in[10];
  const float* b3 = (const float*)d_in[11];
  const float* Wc = (const float*)d_in[12];
  const float* bc = (const float*)d_in[13];
  const float* Wo = (const float*)d_in[14];
  const float* bo = (const float*)d_in[15];
  float* out = (float*)d_out;

  deepfm_fused<<<B_SZ / TILE_M, 256, LDS_BYTES, stream>>>(
      x_cat, x_num, emb0, emb1, num_bias, num_emb,
      W1, b1, W2, b2, W3, b3, Wc, bc, Wo, bo, out);
}
